// RelativeMultiHeadAttenion_68496138436889
// MI455X (gfx1250) — compile-verified, matrix operand order checked by reading, not run on hardware
//
#include <hip/hip_runtime.h>

typedef unsigned short u16;
typedef unsigned int   u32;
typedef __attribute__((ext_vector_type(16))) __bf16 bf16x16;
typedef __attribute__((ext_vector_type(8)))  float  f32x8;

// fp32 -> bf16, round-to-nearest-even
__device__ __forceinline__ u16 f2bf(float f) {
  u32 u = __float_as_uint(f);
  u32 r = u + 0x7FFFu + ((u >> 16) & 1u);
  return (u16)(r >> 16);
}

union Frag {
  bf16x16 v;
  uint4   q[2];
};

__device__ __forceinline__ f32x8 wmma_bf16(const Frag& a, const Frag& b, f32x8 c) {
  // D = A(16x32 bf16) * B(32x16 bf16) + C(16x16 f32)
  return __builtin_amdgcn_wmma_f32_16x16x32_bf16(false, a.v, false, b.v,
                                                 (short)0, c, false, false);
}

__device__ __forceinline__ void zero_acc(f32x8& a) {
#pragma unroll
  for (int e = 0; e < 8; ++e) a[e] = 0.0f;
}

// ---------------------------------------------------------------- converts

__global__ void k_cvt(const float* __restrict__ s, u16* __restrict__ d, int n) {
  int i = blockIdx.x * blockDim.x + threadIdx.x;
  if (i < n) d[i] = f2bf(s[i]);
}

// dst[c*R + r] = bf16(src[r*C + c])  (transpose while converting, for B operands)
__global__ void k_cvt_tr(const float* __restrict__ s, u16* __restrict__ d, int R, int C) {
  int i = blockIdx.x * blockDim.x + threadIdx.x;
  if (i < R * C) {
    int r = i / C, c = i % C;
    d[c * R + r] = f2bf(s[i]);
  }
}

// ---------------------------------------------------------------- GEMM
// C[M,N] = A[M,K](bf16,row-major) * Bt[N,K](bf16: B pre-transposed) ; fp32 acc.
// mode 0: store bf16 row-major [M,N]
// mode 1: store bf16 with per-head K-transpose remap -> Ktr[B,H,L,d]
// mode 2: store bf16 with per-head V-transpose remap -> Vtr[B,H,d,L]
// mode 3: store fp32 row-major [M,N]
__global__ __launch_bounds__(256) void k_gemm(const u16* __restrict__ A,
                                              const u16* __restrict__ Bt,
                                              u16* __restrict__ obf,
                                              float* __restrict__ of,
                                              int M, int N, int K, int mode) {
  const int bm = blockIdx.x * 128, bn = blockIdx.y * 128;
  const int tid = threadIdx.x, wave = tid >> 5, lane = tid & 31;
  const int lo = lane & 15, hi = lane >> 4;
  const int wr = (wave & 3) * 32;   // 4 wave-rows  x 32 rows
  const int wc = (wave >> 2) * 64;  // 2 wave-cols  x 64 cols

  f32x8 acc[2][4];
#pragma unroll
  for (int rt = 0; rt < 2; ++rt)
#pragma unroll
    for (int ct = 0; ct < 4; ++ct) zero_acc(acc[rt][ct]);

  for (int k0 = 0; k0 < K; k0 += 32) {
    Frag fa[2], fb[4];
#pragma unroll
    for (int rt = 0; rt < 2; ++rt) {
      const u16* ap = A + (size_t)(bm + wr + rt * 16 + lo) * K + k0 + hi * 8;
      fa[rt].q[0] = *(const uint4*)ap;
      fa[rt].q[1] = *(const uint4*)(ap + 16);
    }
#pragma unroll
    for (int ct = 0; ct < 4; ++ct) {
      const u16* bp = Bt + (size_t)(bn + wc + ct * 16 + lo) * K + k0 + hi * 8;
      fb[ct].q[0] = *(const uint4*)bp;
      fb[ct].q[1] = *(const uint4*)(bp + 16);
    }
    // hint A k-slab 4 iterations ahead into cache (global_prefetch_b8)
    __builtin_prefetch(A + (size_t)(bm + wr + lo) * K + k0 + 128, 0, 0);
#pragma unroll
    for (int rt = 0; rt < 2; ++rt)
#pragma unroll
      for (int ct = 0; ct < 4; ++ct)
        acc[rt][ct] = wmma_bf16(fa[rt], fb[ct], acc[rt][ct]);
  }

#pragma unroll
  for (int rt = 0; rt < 2; ++rt) {
#pragma unroll
    for (int ct = 0; ct < 4; ++ct) {
#pragma unroll
      for (int e = 0; e < 8; ++e) {
        const int r = bm + wr + rt * 16 + hi * 8 + e;
        const int c = bn + wc + ct * 16 + lo;
        const float val = acc[rt][ct][e];
        if (mode == 0) {
          obf[(size_t)r * N + c] = f2bf(val);
        } else if (mode == 1) {
          // src element = KW[b, l0=h*64+dd, c=l]  -> Ktr[b,h,l,dd]
          const int bb = r >> 10, l0 = r & 1023;
          const int hh = l0 >> 6, dd = l0 & 63;
          obf[((size_t)(bb * 16 + hh) * 1024 + c) * 64 + dd] = f2bf(val);
        } else if (mode == 2) {
          // src element = VW[b, l0, c]; flat=l0*1024+c -> (h,l,dd); Vtr[b,h,dd,l]
          const int bb = r >> 10, l0 = r & 1023;
          const int flat = l0 * 1024 + c;
          const int hh = flat >> 16, rem = flat & 65535;
          const int ll = rem >> 6, dd = rem & 63;
          obf[((size_t)(bb * 16 + hh) * 64 + dd) * 1024 + ll] = f2bf(val);
        } else {
          of[(size_t)r * N + c] = val;
        }
      }
    }
  }
}

// ---------------------------------------------------------------- attention
// One workgroup per (b, h, 32-row block). 256 threads = 8 waves.
// LDS: S[32][1024] f32 (128KB) + P[32][1024] bf16 (64KB) = 192KB dynamic.
__global__ __launch_bounds__(256) void k_attn(const u16* __restrict__ Q,
                                              const u16* __restrict__ Ktr,
                                              const u16* __restrict__ Vtr,
                                              const u16* __restrict__ Er,
                                              u16* __restrict__ AO) {
  extern __shared__ char smem[];
  float* S = (float*)smem;                  // [32][1024]
  u16*   P = (u16*)(smem + 32 * 1024 * 4);  // [32][1024]

  const int blk = blockIdx.x;
  const int ib = blk & 31, h = (blk >> 5) & 15, b = blk >> 9;
  const int i0 = ib * 32;
  const int tid = threadIdx.x, wave = tid >> 5, lane = tid & 31;
  const int lo = lane & 15, hi = lane >> 4;

  const u16* Qh = Q   + (size_t)b * 1048576 + (size_t)h * 65536;  // [1024][64]
  const u16* Kh = Ktr + (size_t)(b * 16 + h) * 65536;             // [1024][64] (l,dd)
  const u16* Vh = Vtr + (size_t)(b * 16 + h) * 65536;             // [64][1024] (dd,l)

  for (int i = tid; i < 32 * 1024; i += 256) S[i] = 0.0f;
  __syncthreads();

  // ---- Srel: compute QEr tiles (rows i0..i0+47, guarded) and scatter-skew into S
  for (int t = wave; t < 192; t += 8) {
    const int rt = t >> 6, ct = t & 63;
    const int i2b = i0 + rt * 16, cb = ct * 16;
    f32x8 acc; zero_acc(acc);
#pragma unroll
    for (int ks = 0; ks < 2; ++ks) {
      const int k0 = ks * 32;
      Frag fa, fb;
      const int row = i2b + lo;
      if (row < 1024) {
        const u16* ap = Qh + (size_t)row * 64 + k0 + hi * 8;
        fa.q[0] = *(const uint4*)ap;
        fa.q[1] = *(const uint4*)(ap + 16);
      } else {
        fa.q[0] = make_uint4(0u, 0u, 0u, 0u);
        fa.q[1] = make_uint4(0u, 0u, 0u, 0u);
      }
      const u16* bp = Er + (size_t)(cb + lo) * 64 + k0 + hi * 8;
      fb.q[0] = *(const uint4*)bp;
      fb.q[1] = *(const uint4*)(bp + 16);
      acc = wmma_bf16(fa, fb, acc);
    }
#pragma unroll
    for (int e = 0; e < 8; ++e) {
      const int i2 = i2b + hi * 8 + e;
      if (i2 > 1023) continue;
      const int c = cb + lo;
      int irow, j;
      if (c >= 1023 - i2) { irow = i2;     j = c + i2 - 1023; }  // lower: Srel[i, c+i-1023]
      else               { irow = i2 - 1; j = c + i2 + 1;    }  // upper: Srel[i2-1, c+i2+1]
      if (irow >= i0 && irow < i0 + 32) S[(irow - i0) * 1024 + j] = acc[e];
    }
  }
  __syncthreads();

  // ---- S += Q @ K^T
  for (int t = wave; t < 128; t += 8) {
    const int rt = t >> 6, ct = t & 63;
    const int rb = i0 + rt * 16, cb = ct * 16;
    f32x8 acc; zero_acc(acc);
#pragma unroll
    for (int ks = 0; ks < 2; ++ks) {
      const int k0 = ks * 32;
      Frag fa, fb;
      const u16* ap = Qh + (size_t)(rb + lo) * 64 + k0 + hi * 8;
      fa.q[0] = *(const uint4*)ap;
      fa.q[1] = *(const uint4*)(ap + 16);
      const u16* bp = Kh + (size_t)(cb + lo) * 64 + k0 + hi * 8;
      fb.q[0] = *(const uint4*)bp;
      fb.q[1] = *(const uint4*)(bp + 16);
      acc = wmma_bf16(fa, fb, acc);
    }
#pragma unroll
    for (int e = 0; e < 8; ++e)
      S[(rb - i0 + hi * 8 + e) * 1024 + cb + lo] += acc[e];
  }
  __syncthreads();

  // ---- softmax over rows (scale 1/sqrt(64)=0.125). One wave per row, shfl reductions.
  for (int rr = 0; rr < 4; ++rr) {
    const int row = wave * 4 + rr;
    float* sr = S + row * 1024;
    float m = -3.402823466e38f;
    for (int c = lane; c < 1024; c += 32) m = fmaxf(m, sr[c] * 0.125f);
#pragma unroll
    for (int off = 16; off > 0; off >>= 1) m = fmaxf(m, __shfl_xor(m, off, 32));
    float sum = 0.0f;
    for (int c = lane; c < 1024; c += 32) {
      const float p = __expf(sr[c] * 0.125f - m);
      sr[c] = p;
      sum += p;
    }
#pragma unroll
    for (int off = 16; off > 0; off >>= 1) sum += __shfl_xor(sum, off, 32);
    const float inv = 1.0f / sum;
    for (int c = lane; c < 1024; c += 32) P[row * 1024 + c] = f2bf(sr[c] * inv);
  }
  __syncthreads();

  // ---- O[32,64] = P[32,1024] @ V[1024,64] ; one 16x16 tile per wave
  {
    const int rt = wave & 1, nt = wave >> 1;
    f32x8 acc; zero_acc(acc);
    for (int ks = 0; ks < 32; ++ks) {
      const int k0 = ks * 32;
      Frag fa, fb;
      const u16* ap = P + (rt * 16 + lo) * 1024 + k0 + hi * 8;   // LDS reads
      fa.q[0] = *(const uint4*)ap;
      fa.q[1] = *(const uint4*)(ap + 16);
      const u16* bp = Vh + (size_t)(nt * 16 + lo) * 1024 + k0 + hi * 8;
      fb.q[0] = *(const uint4*)bp;
      fb.q[1] = *(const uint4*)(bp + 16);
      acc = wmma_bf16(fa, fb, acc);
    }
#pragma unroll
    for (int e = 0; e < 8; ++e) {
      const int l = i0 + rt * 16 + hi * 8 + e;
      const int col = h * 64 + nt * 16 + lo;            // out3[b,l,h*64+dd]
      AO[((size_t)b * 1024 + l) * 1024 + col] = f2bf(acc[e]);
    }
  }
}

// ---------------------------------------------------------------- launch

extern "C" void kernel_launch(void* const* d_in, const int* in_sizes, int n_in,
                              void* d_out, int out_size, void* d_ws, size_t ws_size,
                              hipStream_t stream) {
  (void)in_sizes; (void)n_in; (void)out_size; (void)ws_size;
  const float* q   = (const float*)d_in[0];
  const float* kk  = (const float*)d_in[1];
  const float* vv  = (const float*)d_in[2];
  const float* WQ  = (const float*)d_in[3];
  const float* WK  = (const float*)d_in[4];
  const float* WV  = (const float*)d_in[5];
  const float* Er  = (const float*)d_in[6];
  const float* WM  = (const float*)d_in[7];
  float* out = (float*)d_out;

  const size_t SX = 4u * 1024u * 1024u;  // 4,194,304 elems per activation
  const size_t SW = 1024u * 1024u;       // per weight
  u16* ws   = (u16*)d_ws;
  u16* qbf  = ws;
  u16* kbf  = qbf + SX;
  u16* vbf  = kbf + SX;
  u16* wqT  = vbf + SX;
  u16* wkT  = wqT + SW;
  u16* wvT  = wkT + SW;
  u16* wmT  = wvT + SW;
  u16* erbf = wmT + SW;
  u16* QW   = erbf + 65536;
  u16* KT   = QW + SX;
  u16* VT   = KT + SX;
  u16* AO   = VT + SX;

  const dim3 blk(256);

  k_cvt<<<(int)((SX + 255) / 256), blk, 0, stream>>>(q,  qbf, (int)SX);
  k_cvt<<<(int)((SX + 255) / 256), blk, 0, stream>>>(kk, kbf, (int)SX);
  k_cvt<<<(int)((SX + 255) / 256), blk, 0, stream>>>(vv, vbf, (int)SX);
  k_cvt_tr<<<(int)((SW + 255) / 256), blk, 0, stream>>>(WQ, wqT, 1024, 1024);
  k_cvt_tr<<<(int)((SW + 255) / 256), blk, 0, stream>>>(WK, wkT, 1024, 1024);
  k_cvt_tr<<<(int)((SW + 255) / 256), blk, 0, stream>>>(WV, wvT, 1024, 1024);
  k_cvt_tr<<<(int)((SW + 255) / 256), blk, 0, stream>>>(WM, wmT, 1024, 1024);
  k_cvt<<<65536 / 256, blk, 0, stream>>>(Er, erbf, 65536);

  const dim3 gg(4096 / 128, 1024 / 128);  // 32 x 8 blocks
  k_gemm<<<gg, blk, 0, stream>>>(qbf, wqT, QW, nullptr, 4096, 1024, 1024, 0);
  k_gemm<<<gg, blk, 0, stream>>>(kbf, wkT, KT, nullptr, 4096, 1024, 1024, 1);
  k_gemm<<<gg, blk, 0, stream>>>(vbf, wvT, VT, nullptr, 4096, 1024, 1024, 2);

  k_attn<<<2048, blk, 196608, stream>>>(QW, KT, VT, erbf, AO);

  k_gemm<<<gg, blk, 0, stream>>>(AO, wmT, nullptr, out, 4096, 1024, 1024, 3);
}
